// Label_Attention_3624952398168
// MI455X (gfx1250) — matrix operand order repaired and checked by reading, not compiled, not run on hardware
//
#include <hip/hip_runtime.h>
#include <stdint.h>

// ---------------------------------------------------------------------------
// Label attention:  out = softmax(X @ W^T) @ W   (per head: intent N=512, slot N=1024)
// X: [B*S, 256] f32, W: [N, 256] f32, out: [B*S, 256] f32.
// Flash-style fused kernel, bf16 WMMA (v_wmma_f32_16x16x32_bf16), f32 softmax.
// One wave = one 16-token tile. 8 waves / workgroup, W chunk staged in LDS.
// ---------------------------------------------------------------------------

typedef __attribute__((ext_vector_type(16))) __bf16 v16bf;
typedef __attribute__((ext_vector_type(8)))  float  v8f;

union FragBF {
    uint4   q[2];
    v16bf   v;
};

#define DD 256

// round-to-nearest-even f32 -> bf16 (bit pattern)
__device__ inline unsigned int f2bf(float f) {
    union { float f; unsigned int u; } c; c.f = f;
    unsigned int x = c.u;
    return (x + 0x7FFFu + ((x >> 16) & 1u)) >> 16;
}
__device__ inline unsigned int pack2bf(float a, float b) {
    return f2bf(a) | (f2bf(b) << 16);
}

__global__ __launch_bounds__(256, 1)
void label_attn_kernel(const float* __restrict__ X,
                       const float* __restrict__ W,
                       float* __restrict__ out,
                       int N)
{
    extern __shared__ char smem[];
    // LDS layout:
    //   [0,16384)          : Wbf  bf16 [32 labels][256 d]      (B of scores GEMM)
    //   [16384,32768)      : Wt   bf16 [256 d][32 labels]      (B of probs GEMM)
    //   [32768, +8*8192)   : per-wave Xbf  bf16 [16][256]
    //   [98304, +8*1024)   : per-wave Ptile bf16 [16][32]
    char* Wbf = smem;
    char* Wt  = smem + 16384;
    const int wave = threadIdx.x >> 5;
    const int lane = threadIdx.x & 31;
    char* Xbf   = smem + 32768 + wave * 8192;
    char* Ptile = smem + 32768 + 8 * 8192 + wave * 1024;

    const int  tile      = blockIdx.x * 8 + wave;        // 16-token tile id
    const long tokenBase = (long)tile * 16;
    const int  hi = lane >> 4;                           // which 16-lane half
    const int  ln = lane & 15;
    const int  nChunks = N >> 5;                         // 32 labels per chunk

    // ---- stage X tile -> bf16 LDS (rows are contiguous: 4096 f32 region) ----
    {
        const float4* xg = (const float4*)(X + tokenBase * DD);
        uint2* xs = (uint2*)Xbf;
        #pragma unroll
        for (int i = 0; i < 32; ++i) {
            int idx4 = lane + i * 32;                    // 0..1023 float4s
            float4 f = xg[idx4];
            uint2 p; p.x = pack2bf(f.x, f.y); p.y = pack2bf(f.z, f.w);
            xs[idx4] = p;
        }
    }
    asm volatile("" ::: "memory");

    // ---- cooperative W-chunk staging (whole workgroup) ----
    // Pass A: row-major coalesced float4 reads -> Wbf (conflict-free b64 stores).
    // Pass B (only when the probs GEMM needs W^T): label-major mapping, so the
    //        32 lanes write 2B each into a contiguous 64B span of one Wt row
    //        -> conflict-free (the old dcol-major mapping put all 32 lanes on
    //        one bank: stride 4 rows * 64B = 64 DWORDs = same bank, 32-way).
    auto stageW = [&](int c, bool withT) {
        __syncthreads();                                 // previous chunk consumed
        {
            const float4* wg = (const float4*)(W + (long)c * 32 * DD);
            uint2* wbf = (uint2*)Wbf;
            #pragma unroll
            for (int i = 0; i < 8; ++i) {
                int idx4 = threadIdx.x + i * 256;        // 0..2047 float4s
                float4 f = wg[idx4];
                uint2 p; p.x = pack2bf(f.x, f.y); p.y = pack2bf(f.z, f.w);
                wbf[idx4] = p;                           // row-major bf16
            }
        }
        if (withT) {
            const int l  = threadIdx.x & 31;             // label within chunk
            const int dg = threadIdx.x >> 5;             // 0..7
            const float4* wr = (const float4*)(W + ((long)c * 32 + l) * DD);
            unsigned short* wt = (unsigned short*)Wt;
            #pragma unroll
            for (int i = 0; i < 8; ++i) {
                int d0 = dg * 4 + i * 32;                // float4 col block
                float4 f = wr[d0 >> 2];                  // L2-resident re-read
                wt[(d0 + 0) * 32 + l] = (unsigned short)f2bf(f.x);
                wt[(d0 + 1) * 32 + l] = (unsigned short)f2bf(f.y);
                wt[(d0 + 2) * 32 + l] = (unsigned short)f2bf(f.z);
                wt[(d0 + 3) * 32 + l] = (unsigned short)f2bf(f.w);
            }
        }
        __syncthreads();
    };

    // ---- one 16x16 score tile: S = Xtile(16x256) @ Wlabels(16x256)^T ----
    // A frag (16x32 bf16): lane<16 holds row ln, K = {k0..k0+7, k0+16..k0+23}
    // B frag (32x16 bf16): lane holds label column ln, K = {k0..k0+7, k0+16..k0+23}
    auto scoreTile = [&](int t) -> v8f {
        v8f acc = {0.f, 0.f, 0.f, 0.f, 0.f, 0.f, 0.f, 0.f};
        #pragma unroll
        for (int kd = 0; kd < 8; ++kd) {                 // K = 256 / 32
            FragBF a, b;
            const uint4* ap = (const uint4*)(Xbf + ln * 512 + kd * 64 + hi * 16);
            a.q[0] = ap[0];
            a.q[1] = ap[2];                              // +32B -> K+16
            const uint4* bp = (const uint4*)(Wbf + (t * 16 + ln) * 512 + kd * 64 + hi * 16);
            b.q[0] = bp[0];
            b.q[1] = bp[2];
            acc = __builtin_amdgcn_wmma_f32_16x16x32_bf16(
                false, a.v, false, b.v, (short)0, acc, false, false);
        }
        return acc;
    };

    // ============ Sweep 1: online softmax statistics (row max, row sum) =====
    float m[8], l[8];
    #pragma unroll
    for (int v = 0; v < 8; ++v) { m[v] = -3.0e38f; l[v] = 0.f; }

    for (int c = 0; c < nChunks; ++c) {
        stageW(c, false);
        if (c + 1 < nChunks)
            __builtin_prefetch((const char*)(W + (long)(c + 1) * 32 * DD) + threadIdx.x * 128, 0, 1);
        #pragma unroll
        for (int t = 0; t < 2; ++t) {
            v8f s = scoreTile(t);
            #pragma unroll
            for (int v = 0; v < 8; ++v) {
                // row (v + 8*hi) lives across the 16 lanes of this half
                float cm = s[v];
                cm = fmaxf(cm, __shfl_xor(cm, 1, 16));
                cm = fmaxf(cm, __shfl_xor(cm, 2, 16));
                cm = fmaxf(cm, __shfl_xor(cm, 4, 16));
                cm = fmaxf(cm, __shfl_xor(cm, 8, 16));
                float mn = fmaxf(m[v], cm);
                float e = __expf(s[v] - mn);
                e += __shfl_xor(e, 1, 16);
                e += __shfl_xor(e, 2, 16);
                e += __shfl_xor(e, 4, 16);
                e += __shfl_xor(e, 8, 16);
                l[v] = l[v] * __expf(m[v] - mn) + e;
                m[v] = mn;
            }
        }
    }
    float rl[8];
    #pragma unroll
    for (int v = 0; v < 8; ++v) rl[v] = 1.0f / l[v];

    // ============ Sweep 2: recompute scores, P = exp(s-m)/l, O += P @ W =====
    v8f O[16];
    #pragma unroll
    for (int j = 0; j < 16; ++j)
        #pragma unroll
        for (int v = 0; v < 8; ++v) O[j][v] = 0.f;

    for (int c = 0; c < nChunks; ++c) {
        stageW(c, true);
        if (c + 1 < nChunks)
            __builtin_prefetch((const char*)(W + (long)(c + 1) * 32 * DD) + threadIdx.x * 128, 0, 1);
        unsigned short* pt = (unsigned short*)Ptile;
        #pragma unroll
        for (int t = 0; t < 2; ++t) {
            v8f s = scoreTile(t);
            #pragma unroll
            for (int v = 0; v < 8; ++v) {
                float p = __expf(s[v] - m[v]) * rl[v];
                // C-layout element (v,lane) = row v+8*hi, col t*16+ln
                pt[(v + 8 * hi) * 32 + t * 16 + ln] = (unsigned short)f2bf(p);
            }
        }
        // per-wave LDS RAW: DS ops are in-order per wave; make it explicit
        asm volatile("s_wait_dscnt 0x0" ::: "memory");

        // A frag from Ptile (16 rows x 32 labels, bf16, row stride 64B)
        FragBF a;
        const uint4* ap = (const uint4*)(Ptile + ln * 64 + hi * 16);
        a.q[0] = ap[0];
        a.q[1] = ap[2];
        #pragma unroll
        for (int j = 0; j < 16; ++j) {                   // 16 output col tiles
            FragBF b;
            const uint4* bp = (const uint4*)(Wt + (j * 16 + ln) * 64 + hi * 16);
            b.q[0] = bp[0];
            b.q[1] = bp[2];
            O[j] = __builtin_amdgcn_wmma_f32_16x16x32_bf16(
                false, a.v, false, b.v, (short)0, O[j], false, false);
        }
    }

    // ---- write O (16 tokens x 256) ----
    float* og = out + tokenBase * DD;
    #pragma unroll
    for (int j = 0; j < 16; ++j)
        #pragma unroll
        for (int v = 0; v < 8; ++v)
            og[(v + 8 * hi) * DD + j * 16 + ln] = O[j][v];
}

extern "C" void kernel_launch(void* const* d_in, const int* in_sizes, int n_in,
                              void* d_out, int out_size, void* d_ws, size_t ws_size,
                              hipStream_t stream) {
    const float* x_intent = (const float*)d_in[0];
    const float* x_slot   = (const float*)d_in[1];
    // d_in[2] = mask (unused by reference math)
    const float* W_intent = (const float*)d_in[3];
    const float* W_slot   = (const float*)d_in[4];
    float* out = (float*)d_out;

    const int tokens  = in_sizes[0] / 256;   // B*S = 32768
    const int tiles   = tokens / 16;         // 2048
    const int blocks  = tiles / 8;           // 256 (8 waves/WG, 1 tile/wave)
    const int n_int   = in_sizes[3] / 256;   // 512
    const int n_slot  = in_sizes[4] / 256;   // 1024
    const size_t smemBytes = 32768 + 8 * 8192 + 8 * 1024;  // 106,496 B

    label_attn_kernel<<<blocks, 256, smemBytes, stream>>>(
        x_intent, W_intent, out, n_int);
    label_attn_kernel<<<blocks, 256, smemBytes, stream>>>(
        x_slot, W_slot, out + (size_t)tokens * 256, n_slot);
}
